// TransformerModel_77326591197706
// MI455X (gfx1250) — compile-verified, matrix operand order checked by reading, not run on hardware
//
#include <hip/hip_runtime.h>
#include <cmath>

// ---- model dims ----
static constexpr int NB = 4;      // batch
static constexpr int NS = 1024;   // seq
static constexpr int ND = 1024;   // d_model
static constexpr int NH = 16;     // heads
static constexpr int NL = 12;     // layers
static constexpr int NV = 32000;  // vocab
static constexpr int NDH = 64;    // head dim
static constexpr int NM = NB * NS; // 4096 rows

// ---- types ----
typedef __attribute__((ext_vector_type(16))) __bf16 bf16x16;
typedef __attribute__((ext_vector_type(8)))  __bf16 bf16x8;
typedef __attribute__((ext_vector_type(8)))  float  f32x8;
typedef int v4i_gcc __attribute__((vector_size(16)));
typedef __attribute__((address_space(1))) v4i_gcc* gas_b128_t;
typedef __attribute__((address_space(3))) v4i_gcc* las_b128_t;

union Frag { bf16x16 v; bf16x8 h[2]; };

__device__ __forceinline__ unsigned short f2bfu(float f) {
    unsigned u = __builtin_bit_cast(unsigned, f);
    u += 0x7FFFu + ((u >> 16) & 1u);            // round-to-nearest-even
    return (unsigned short)(u >> 16);
}
__device__ __forceinline__ unsigned long long pack4(float a, float b, float c, float d) {
    return (unsigned long long)f2bfu(a) |
           ((unsigned long long)f2bfu(b) << 16) |
           ((unsigned long long)f2bfu(c) << 32) |
           ((unsigned long long)f2bfu(d) << 48);
}
__device__ __forceinline__ f32x8 wmma_bf16(const Frag& a, const Frag& b, f32x8 c) {
    return __builtin_amdgcn_wmma_f32_16x16x32_bf16(false, a.v, false, b.v, (short)0, c, false, false);
}
__device__ __forceinline__ bf16x8 lds_ld8(const unsigned short* p) {
    return *(const bf16x8*)p;
}

// ---- CDNA5 async global->LDS copy (ASYNCcnt) ----
__device__ __forceinline__ void async_copy_b128(const unsigned short* g, unsigned short* l) {
#if __has_builtin(__builtin_amdgcn_global_load_async_to_lds_b128)
    __builtin_amdgcn_global_load_async_to_lds_b128(
        (gas_b128_t)(void*)g, (las_b128_t)(void*)l, 0, 0);
#else
    unsigned loff = (unsigned)(unsigned long long)(void*)l;   // LDS offset = low 32 bits of pointer
    unsigned long long ga = (unsigned long long)g;
    asm volatile("global_load_async_to_lds_b128 %0, %1, off"
                 :: "v"(loff), "v"(ga) : "memory");
#endif
}
__device__ __forceinline__ void wait_async() {
#if __has_builtin(__builtin_amdgcn_s_wait_asynccnt)
    __builtin_amdgcn_s_wait_asynccnt(0);
#else
    asm volatile("s_wait_asynccnt 0" ::: "memory");
#endif
}

// xor-shuffle within 16-lane halves via single ds_swizzle_b32
template <int MASK>
__device__ __forceinline__ float swz_xor(float x) {
    int i = __builtin_amdgcn_ds_swizzle(__builtin_bit_cast(int, x), (MASK << 10) | 0x1f);
    return __builtin_bit_cast(float, i);
}

// ================= elementwise f32 -> bf16 (wout) =================
__global__ __launch_bounds__(256) void cvt_bf16_kernel(const float* __restrict__ in,
                                                       unsigned short* __restrict__ out,
                                                       size_t n4) {
    size_t stride = (size_t)gridDim.x * blockDim.x;
    for (size_t i = (size_t)blockIdx.x * blockDim.x + threadIdx.x; i < n4; i += stride) {
        float4 v = ((const float4*)in)[i];
        ((unsigned long long*)out)[i] = pack4(v.x, v.y, v.z, v.w);
    }
}

// ========= weight transpose+convert: f32 [K][N] -> bf16 [N][K] (per layer z) =========
__global__ __launch_bounds__(256) void cvt_transpose_kernel(const float* __restrict__ in,
                                                            unsigned short* __restrict__ out,
                                                            int K, int N) {
    __shared__ unsigned short tile[64][72];   // 64x64 tile, pad 8 (keeps 8B alignment)
    size_t loff = (size_t)blockIdx.z * K * N;
    const float* inp = in + loff;
    unsigned short* outp = out + loff;
    int k0 = blockIdx.y * 64, n0 = blockIdx.x * 64;
    int tid = threadIdx.x;
#pragma unroll
    for (int i = 0; i < 4; ++i) {
        int e = tid + 256 * i;               // 1024 vec4: 64 k-rows x 16 n-vec4
        int kr = e >> 4, nv = e & 15;
        float4 v = *(const float4*)&inp[(size_t)(k0 + kr) * N + n0 + nv * 4];
        tile[nv * 4 + 0][kr] = f2bfu(v.x);
        tile[nv * 4 + 1][kr] = f2bfu(v.y);
        tile[nv * 4 + 2][kr] = f2bfu(v.z);
        tile[nv * 4 + 3][kr] = f2bfu(v.w);
    }
    __syncthreads();
#pragma unroll
    for (int i = 0; i < 4; ++i) {
        int e = tid + 256 * i;               // 1024 groups of 4 ushort
        int nr = e >> 4, kv = e & 15;
        unsigned long long pk = *(const unsigned long long*)&tile[nr][kv * 4];
        *(unsigned long long*)&outp[(size_t)(n0 + nr) * K + k0 + kv * 4] = pk;
    }
}

// ================= embedding + positional =================
__global__ __launch_bounds__(256) void embed_kernel(const int* __restrict__ X,
                                                    const float* __restrict__ emb,
                                                    const float* __restrict__ pe,
                                                    float* __restrict__ x) {
    int e  = blockIdx.x * 256 + threadIdx.x;       // vec4 index over [NM * ND/4]
    int d4 = e & 255;                               // ND/4 = 256
    int bs = e >> 8;
    int b  = bs >> 10;                              // NS = 1024
    int tok = X[bs];
    float4 ev = ((const float4*)(emb + (size_t)tok * ND))[d4];
    float4 pv = ((const float4*)(pe  + (size_t)b   * ND))[d4];
    float4 o;
    o.x = ev.x * 32.0f + pv.x;  o.y = ev.y * 32.0f + pv.y;
    o.z = ev.z * 32.0f + pv.z;  o.w = ev.w * 32.0f + pv.w;
    ((float4*)x)[e] = o;
}

// ================= layernorm: writes f32 and bf16 copies =================
__global__ __launch_bounds__(256) void ln_kernel(const float* __restrict__ x,
                                                 const float* __restrict__ g,
                                                 const float* __restrict__ b,
                                                 float* __restrict__ o,
                                                 unsigned short* __restrict__ obf) {
    int row = blockIdx.x;
    int t = threadIdx.x;
    const float* xr = x + (size_t)row * ND;
    float4 v = ((const float4*)xr)[t];
    float s  = v.x + v.y + v.z + v.w;
    float sq = v.x*v.x + v.y*v.y + v.z*v.z + v.w*v.w;
    __shared__ float rs[256], rq[256];
    rs[t] = s; rq[t] = sq;
    __syncthreads();
    for (int off = 128; off > 0; off >>= 1) {
        if (t < off) { rs[t] += rs[t + off]; rq[t] += rq[t + off]; }
        __syncthreads();
    }
    float mu  = rs[0] * (1.0f / ND);
    float var = rq[0] * (1.0f / ND) - mu * mu;
    float inv = rsqrtf(var + 1e-5f);
    float4 gg = ((const float4*)g)[t];
    float4 bb = ((const float4*)b)[t];
    float4 ov;
    ov.x = (v.x - mu) * inv * gg.x + bb.x;
    ov.y = (v.y - mu) * inv * gg.y + bb.y;
    ov.z = (v.z - mu) * inv * gg.z + bb.z;
    ov.w = (v.w - mu) * inv * gg.w + bb.w;
    ((float4*)(o + (size_t)row * ND))[t] = ov;
    *(unsigned long long*)&obf[(size_t)row * ND + t * 4] = pack4(ov.x, ov.y, ov.z, ov.w);
}

// ================= bf16 WMMA GEMM (async double-buffered) =================
// C = A[M][K](bf16) * B[N][K]^T(bf16) + bias + res
// OUT_MODE: 0 = f32 C, 1 = bf16 Cbf only, 2 = both.
// Block: 256 thr = 8 waves; C tile 128x128; wave tile 64x32; K-step 32.
template <bool HAS_BIAS, bool HAS_RES, int OUT_MODE>
__global__ __launch_bounds__(256) void gemm_bf16_kernel(
    const unsigned short* __restrict__ A, const unsigned short* __restrict__ Bw,
    const float* __restrict__ bias, const float* __restrict__ res,
    float* __restrict__ C, unsigned short* __restrict__ Cbf,
    int M, int N, int K) {
    __shared__ __align__(16) unsigned short As[2][128 * 32];   // [m][k]
    __shared__ __align__(16) unsigned short Bs[2][128 * 32];   // [n][k]

    int tid  = threadIdx.x;
    int lane = tid & 31;
    int wave = tid >> 5;
    int wm = wave >> 2;          // 0..1 -> 64 rows each
    int wn = wave & 3;           // 0..3 -> 32 cols each
    int m0 = blockIdx.y * 128;
    int n0 = blockIdx.x * 128;
    int lm  = lane & 15;
    int ks8 = (lane >> 4) * 8;

    f32x8 acc[4][2];
#pragma unroll
    for (int i = 0; i < 4; ++i)
#pragma unroll
        for (int j = 0; j < 2; ++j) acc[i][j] = (f32x8)0.0f;

    auto stage = [&](int buf, int k0) {
#pragma unroll
        for (int i = 0; i < 2; ++i) {
            int e = tid + 256 * i;           // 512 16B segments: row = e>>2, seg = e&3
            int row = e >> 2, seg = e & 3;
            async_copy_b128(&A[(size_t)(m0 + row) * K + k0 + seg * 8],
                            &As[buf][row * 32 + seg * 8]);
        }
#pragma unroll
        for (int i = 0; i < 2; ++i) {
            int e = tid + 256 * i;
            int row = e >> 2, seg = e & 3;
            async_copy_b128(&Bw[(size_t)(n0 + row) * K + k0 + seg * 8],
                            &Bs[buf][row * 32 + seg * 8]);
        }
    };

    const int KT = K >> 5;
    stage(0, 0);
    wait_async();
    __syncthreads();

    for (int kt = 0; kt < KT; ++kt) {
        int cur = kt & 1;
        if (kt + 1 < KT) stage(cur ^ 1, (kt + 1) << 5);   // overlap DMA with WMMA

        Frag af[4], bfr[2];
#pragma unroll
        for (int mt = 0; mt < 4; ++mt) {
            const unsigned short* p = &As[cur][(wm * 64 + mt * 16 + lm) * 32 + ks8];
            af[mt].h[0] = lds_ld8(p);
            af[mt].h[1] = lds_ld8(p + 16);
        }
#pragma unroll
        for (int nt = 0; nt < 2; ++nt) {
            const unsigned short* p = &Bs[cur][(wn * 32 + nt * 16 + lm) * 32 + ks8];
            bfr[nt].h[0] = lds_ld8(p);
            bfr[nt].h[1] = lds_ld8(p + 16);
        }
#pragma unroll
        for (int mt = 0; mt < 4; ++mt)
#pragma unroll
            for (int nt = 0; nt < 2; ++nt)
                acc[mt][nt] = wmma_bf16(af[mt], bfr[nt], acc[mt][nt]);

        wait_async();
        __syncthreads();
    }

    // epilogue
#pragma unroll
    for (int mt = 0; mt < 4; ++mt)
#pragma unroll
        for (int nt = 0; nt < 2; ++nt) {
            int col = n0 + wn * 32 + nt * 16 + lm;
            float bv = HAS_BIAS ? bias[col] : 0.0f;
#pragma unroll
            for (int i = 0; i < 8; ++i) {
                int row = m0 + wm * 64 + mt * 16 + i + 8 * (lane >> 4);
                float v = acc[mt][nt][i] + bv;
                if (HAS_RES) v += res[(size_t)row * N + col];
                if (OUT_MODE != 1) C[(size_t)row * N + col] = v;
                if (OUT_MODE >= 1) Cbf[(size_t)row * N + col] = f2bfu(v);
            }
        }
}

// ================= flash attention (bf16 qkv in, bf16 o out) =================
// qkv: [NM][3*ND] bf16 (q | k | v). Block: 128 thr = 4 waves, 64 q rows per block,
// one (b, head) per block; online softmax over key chunks of 32.
__global__ __launch_bounds__(128) void attention_kernel(const unsigned short* __restrict__ qkv,
                                                        unsigned short* __restrict__ obf) {
    __shared__ __align__(16) unsigned short Qs[64 * 64];     // [qrow][dh]
    __shared__ __align__(16) unsigned short Ks[32 * 64];     // [key][dh]
    __shared__ __align__(16) unsigned short Vs[64 * 32];     // [dh][key]
    __shared__ __align__(16) unsigned short Ps[4][16 * 32];  // per-wave P tile

    int tid  = threadIdx.x;
    int lane = tid & 31;
    int wave = tid >> 5;
    int lm   = lane & 15;
    int ks8  = (lane >> 4) * 8;

    int qb = blockIdx.x & 15;            // q block (S/64 = 16)
    int hh = (blockIdx.x >> 4) & 15;     // head
    int bb = blockIdx.x >> 8;            // batch
    int rowbase = bb * NS + qb * 64;

    const int LDQ = 3 * ND;
    const unsigned short* qbase = qkv + (size_t)rowbase * LDQ + hh * NDH;
    const unsigned short* kbase = qkv + (size_t)(bb * NS) * LDQ + ND + hh * NDH;
    const unsigned short* vbase = qkv + (size_t)(bb * NS) * LDQ + 2 * ND + hh * NDH;

    // stage Q via async DMA: 64 rows x 8 16B-segments
#pragma unroll
    for (int i = 0; i < 4; ++i) {
        int e = tid + 128 * i;           // 512 segments
        int row = e >> 3, seg = e & 7;
        async_copy_b128(&qbase[(size_t)row * LDQ + seg * 8], &Qs[row * 64 + seg * 8]);
    }
    wait_async();
    __syncthreads();

    Frag qf[2];
    {
        const unsigned short* p = &Qs[(wave * 16 + lm) * 64];
        qf[0].h[0] = lds_ld8(p + ks8);
        qf[0].h[1] = lds_ld8(p + 16 + ks8);
        qf[1].h[0] = lds_ld8(p + 32 + ks8);
        qf[1].h[1] = lds_ld8(p + 48 + ks8);
    }

    f32x8 oacc[4];
#pragma unroll
    for (int t = 0; t < 4; ++t) oacc[t] = (f32x8)0.0f;
    float m_run[8], l_run[8];
#pragma unroll
    for (int i = 0; i < 8; ++i) { m_run[i] = -1e30f; l_run[i] = 0.0f; }

    for (int kb = 0; kb < NS / 32; ++kb) {
        // K chunk [32 keys][64 dh] via async DMA
#pragma unroll
        for (int i = 0; i < 2; ++i) {
            int e = tid + 128 * i;       // 256 segments
            int row = e >> 3, seg = e & 7;
            async_copy_b128(&kbase[(size_t)(kb * 32 + row) * LDQ + seg * 8],
                            &Ks[row * 64 + seg * 8]);
        }
        // V chunk transposed -> Vs[dh][key]
#pragma unroll
        for (int i = 0; i < 4; ++i) {
            int e = tid + 128 * i;       // 512 groups of 4 ushort
            int key = e >> 4, kv = e & 15;
            unsigned long long pk =
                *(const unsigned long long*)&vbase[(size_t)(kb * 32 + key) * LDQ + kv * 4];
            Vs[(kv * 4 + 0) * 32 + key] = (unsigned short)(pk);
            Vs[(kv * 4 + 1) * 32 + key] = (unsigned short)(pk >> 16);
            Vs[(kv * 4 + 2) * 32 + key] = (unsigned short)(pk >> 32);
            Vs[(kv * 4 + 3) * 32 + key] = (unsigned short)(pk >> 48);
        }
        wait_async();
        __syncthreads();

        // scores: 16 q-rows x 32 keys (no 1/sqrt(d) scale, per reference)
        f32x8 s0 = (f32x8)0.0f, s1 = (f32x8)0.0f;
#pragma unroll
        for (int kk = 0; kk < 2; ++kk) {
            Frag kf0, kf1;
            const unsigned short* p0 = &Ks[lm * 64 + kk * 32];
            const unsigned short* p1 = &Ks[(16 + lm) * 64 + kk * 32];
            kf0.h[0] = lds_ld8(p0 + ks8);  kf0.h[1] = lds_ld8(p0 + 16 + ks8);
            kf1.h[0] = lds_ld8(p1 + ks8);  kf1.h[1] = lds_ld8(p1 + 16 + ks8);
            s0 = wmma_bf16(qf[kk], kf0, s0);
            s1 = wmma_bf16(qf[kk], kf1, s1);
        }

        // online softmax (row reductions within 16-lane halves via ds_swizzle)
#pragma unroll
        for (int i = 0; i < 8; ++i) {
            float mx = fmaxf(s0[i], s1[i]);
            mx = fmaxf(mx, swz_xor<1>(mx));
            mx = fmaxf(mx, swz_xor<2>(mx));
            mx = fmaxf(mx, swz_xor<4>(mx));
            mx = fmaxf(mx, swz_xor<8>(mx));
            float mnew = fmaxf(m_run[i], mx);
            float corr = __expf(m_run[i] - mnew);
            float p0 = __expf(s0[i] - mnew);
            float p1 = __expf(s1[i] - mnew);
            s0[i] = p0; s1[i] = p1;
            float rsm = p0 + p1;
            rsm += swz_xor<1>(rsm);
            rsm += swz_xor<2>(rsm);
            rsm += swz_xor<4>(rsm);
            rsm += swz_xor<8>(rsm);
            l_run[i] = l_run[i] * corr + rsm;
            m_run[i] = mnew;
#pragma unroll
            for (int t = 0; t < 4; ++t) oacc[t][i] *= corr;
        }

        // C-layout P -> A-layout via LDS
#pragma unroll
        for (int i = 0; i < 8; ++i) {
            int prow = i + 8 * (lane >> 4);
            Ps[wave][prow * 32 + lm]      = f2bfu(s0[i]);
            Ps[wave][prow * 32 + 16 + lm] = f2bfu(s1[i]);
        }
        __syncthreads();

        Frag pf;
        pf.h[0] = lds_ld8(&Ps[wave][lm * 32 + ks8]);
        pf.h[1] = lds_ld8(&Ps[wave][lm * 32 + 16 + ks8]);
#pragma unroll
        for (int t = 0; t < 4; ++t) {
            Frag vf;
            const unsigned short* p = &Vs[(t * 16 + lm) * 32 + ks8];
            vf.h[0] = lds_ld8(p);
            vf.h[1] = lds_ld8(p + 16);
            oacc[t] = wmma_bf16(pf, vf, oacc[t]);
        }
        __syncthreads();
    }

    // write O (bf16, consumed by proj GEMM)
#pragma unroll
    for (int t = 0; t < 4; ++t)
#pragma unroll
        for (int i = 0; i < 8; ++i) {
            int r = wave * 16 + i + 8 * (lane >> 4);
            int dh = t * 16 + lm;
            obf[(size_t)(rowbase + r) * ND + hh * NDH + dh] = f2bfu(oacc[t][i] / l_run[i]);
        }
}

// ================= host side =================
extern "C" void kernel_launch(void* const* d_in, const int* in_sizes, int n_in,
                              void* d_out, int out_size, void* d_ws, size_t ws_size,
                              hipStream_t stream) {
    const int*   X    = (const int*)d_in[0];
    const float* emb  = (const float*)d_in[1];
    const float* pe   = (const float*)d_in[2];
    const float* ln1g = (const float*)d_in[3];
    const float* ln1b = (const float*)d_in[4];
    const float* wa   = (const float*)d_in[5];
    const float* ba   = (const float*)d_in[6];
    const float* wp   = (const float*)d_in[7];
    const float* bp   = (const float*)d_in[8];
    const float* ln2g = (const float*)d_in[9];
    const float* ln2b = (const float*)d_in[10];
    const float* wf   = (const float*)d_in[11];
    const float* bff  = (const float*)d_in[12];
    const float* wout = (const float*)d_in[13];
    float* logits = (float*)d_out;

    char* ws = (char*)d_ws;
    size_t off = 0;
    auto alloc = [&](size_t bytes) -> void* {
        void* p = ws + off;
        off += (bytes + 255) & ~(size_t)255;
        return p;
    };
    const size_t wa_n   = (size_t)NL * ND * 3 * ND;
    const size_t wp_n   = (size_t)NL * ND * ND;
    const size_t wout_n = (size_t)NV * ND;

    unsigned short* wa_t    = (unsigned short*)alloc(wa_n * 2);    // [L][3D][D] bf16
    unsigned short* wp_t    = (unsigned short*)alloc(wp_n * 2);    // [L][D][D]  bf16
    unsigned short* wf_t    = (unsigned short*)alloc(wp_n * 2);    // [L][D][D]  bf16
    unsigned short* wout_bf = (unsigned short*)alloc(wout_n * 2);  // [V][D]     bf16
    float*          xbuf    = (float*)alloc((size_t)NM * ND * 4);
    float*          hbuf    = (float*)alloc((size_t)NM * ND * 4);
    unsigned short* hbf     = (unsigned short*)alloc((size_t)NM * ND * 2);
    unsigned short* qkv_bf  = (unsigned short*)alloc((size_t)NM * 3 * ND * 2);
    unsigned short* obf     = (unsigned short*)alloc((size_t)NM * ND * 2);
    unsigned short* xbf     = (unsigned short*)alloc((size_t)NM * ND * 2);

    // weights: transpose+convert to bf16 [N][K]; wout is already [N][K]
    cvt_transpose_kernel<<<dim3(3 * ND / 64, ND / 64, NL), 256, 0, stream>>>(wa, wa_t, ND, 3 * ND);
    cvt_transpose_kernel<<<dim3(ND / 64, ND / 64, NL), 256, 0, stream>>>(wp, wp_t, ND, ND);
    cvt_transpose_kernel<<<dim3(ND / 64, ND / 64, NL), 256, 0, stream>>>(wf, wf_t, ND, ND);
    cvt_bf16_kernel<<<4096, 256, 0, stream>>>(wout, wout_bf, wout_n / 4);

    // embed + pos
    embed_kernel<<<(NM * ND / 4) / 256, 256, 0, stream>>>(X, emb, pe, xbuf);

    for (int l = 0; l < NL; ++l) {
        // h = LN1(x)  (f32 residual copy + bf16 GEMM-A copy)
        ln_kernel<<<NM, 256, 0, stream>>>(xbuf, ln1g + l * ND, ln1b + l * ND, hbuf, hbf);
        // qkv = h @ wa + ba  (bf16 out; only attention consumes it)
        gemm_bf16_kernel<true, false, 1><<<dim3(3 * ND / 128, NM / 128), 256, 0, stream>>>(
            hbf, wa_t + (size_t)l * ND * 3 * ND, ba + (size_t)l * 3 * ND, nullptr,
            nullptr, qkv_bf, NM, 3 * ND, ND);
        // flash attention -> bf16 O
        attention_kernel<<<NB * NH * (NS / 64), 128, 0, stream>>>(qkv_bf, obf);
        // x = h + o @ wp + bp
        gemm_bf16_kernel<true, true, 0><<<dim3(ND / 128, NM / 128), 256, 0, stream>>>(
            obf, wp_t + (size_t)l * ND * ND, bp + (size_t)l * ND, hbuf,
            xbuf, nullptr, NM, ND, ND);
        // h = LN2(x)
        ln_kernel<<<NM, 256, 0, stream>>>(xbuf, ln2g + l * ND, ln2b + l * ND, hbuf, hbf);
        // x = x + h @ wf + bf   (last layer also emits bf16 x for the logits GEMM)
        if (l == NL - 1)
            gemm_bf16_kernel<true, true, 2><<<dim3(ND / 128, NM / 128), 256, 0, stream>>>(
                hbf, wf_t + (size_t)l * ND * ND, bff + (size_t)l * ND, xbuf,
                xbuf, xbf, NM, ND, ND);
        else
            gemm_bf16_kernel<true, true, 0><<<dim3(ND / 128, NM / 128), 256, 0, stream>>>(
                hbf, wf_t + (size_t)l * ND * ND, bff + (size_t)l * ND, xbuf,
                xbuf, nullptr, NM, ND, ND);
    }

    // logits = x @ wout^T
    gemm_bf16_kernel<false, false, 0><<<dim3(NV / 128, NM / 128), 256, 0, stream>>>(
        xbf, wout_bf, nullptr, nullptr, logits, nullptr, NM, NV, ND);

    (void)in_sizes; (void)n_in; (void)out_size; (void)ws_size;
}
